// Feature_Align_S_20237885898938
// MI455X (gfx1250) — compile-verified, matrix operand order checked by reading, not run on hardware
//
#include <hip/hip_runtime.h>
#include <hip/hip_bf16.h>

typedef __attribute__((ext_vector_type(16))) _Float16 v16h;
typedef __attribute__((ext_vector_type(8)))  _Float16 v8h;
typedef __attribute__((ext_vector_type(8)))  float    v8f;

#define NB 8
#define NC 128
#define HW 2304
#define CV 16
#define NS 3
#define IT 64      // rows per block (4 waves x 16)
#define JT 32      // J chunk (K of second GEMM, 2 x 16 N-subtiles of first)
#define NWAVE 4
#define VOFF 12288 // half-offset of the vT region inside a tile buffer

// ---------------------------------------------------------------------------
// prep: xb[s][b][j][c] (f16, row-major over c)  <- transpose of data (b,c,hw)
//       sq[s][b][j] = sum_c x^2
//       vT[s][b][o][j] = value_w @ x + value_b   (f16, cv-major for B frags)
// ---------------------------------------------------------------------------
__global__ __launch_bounds__(256) void fas_prep(
    const float* __restrict__ d0, const float* __restrict__ d1,
    const float* __restrict__ d2,
    const float* __restrict__ vw, const float* __restrict__ vb,
    _Float16* __restrict__ xb, _Float16* __restrict__ vT,
    float* __restrict__ sq)
{
  int tid = blockIdx.x * 256 + threadIdx.x;      // tid = ((s*NB+b)*HW + j)
  if (tid >= NS * NB * HW) return;
  int j  = tid % HW;
  int sb = tid / HW;
  int s  = sb / NB;
  const float* xd = (s == 0 ? d0 : (s == 1 ? d1 : d2)) +
                    (size_t)(sb % NB) * NC * HW + j;
  float vacc[CV];
  #pragma unroll
  for (int o = 0; o < CV; ++o) vacc[o] = vb[o];
  float sqv = 0.f;
  _Float16* xrow = xb + (size_t)tid * NC;
  #pragma unroll 8
  for (int c = 0; c < NC; ++c) {
    float xv = xd[(size_t)c * HW];
    sqv = fmaf(xv, xv, sqv);
    xrow[c] = (_Float16)xv;
    #pragma unroll
    for (int o = 0; o < CV; ++o) vacc[o] = fmaf(vw[o * NC + c], xv, vacc[o]);
  }
  sq[tid] = sqv;
  #pragma unroll
  for (int o = 0; o < CV; ++o)
    vT[((size_t)sb * CV + o) * HW + j] = (_Float16)vacc[o];
}

// ---------------------------------------------------------------------------
// CDNA5 async memory->LDS copies (ASYNCcnt), ISA §15.18.3 / §10.7.
// GV addressing: vdst = LDS byte address, vaddr = 64-bit global address.
// ---------------------------------------------------------------------------
__device__ __forceinline__ void fas_async_b128(unsigned lds, const void* g) {
  asm volatile("global_load_async_to_lds_b128 %0, %1, off"
               :: "v"(lds), "v"(g) : "memory");
}
__device__ __forceinline__ void fas_async_b64(unsigned lds, const void* g) {
  asm volatile("global_load_async_to_lds_b64 %0, %1, off"
               :: "v"(lds), "v"(g) : "memory");
}
__device__ __forceinline__ void fas_wait_async_prev() {  // prev tile done,
  asm volatile("s_wait_asynccnt 0xE" ::: "memory");      // next (14 ops) in flight
}
__device__ __forceinline__ void fas_wait_async_all() {
  asm volatile("s_wait_asynccnt 0x0" ::: "memory");
}
__device__ __forceinline__ unsigned fas_lds_addr(const void* p) {
  return (unsigned)(unsigned long long)p;  // LDS aperture: offset in bits[31:0]
}

// Stage one J tile (3 streams x 32 xb rows + 3 x 16 vT rows) into `dst`.
// Exactly 14 async instructions per thread -> uniform per-wave ASYNCcnt.
__device__ __forceinline__ void fas_issue_tile(
    _Float16* dst, const _Float16* __restrict__ xb,
    const _Float16* __restrict__ vT, int batch, int J0, int tid)
{
  #pragma unroll
  for (int i = 0; i < 12; ++i) {               // xb region: 1536 x 16B chunks
    int cidx = i * 128 + tid;
    int s = cidx >> 9, rem = cidx & 511;
    int row = rem >> 4, ch = rem & 15;
    const _Float16* src =
        xb + ((size_t)(s * NB + batch) * HW + J0 + row) * NC + ch * 8;
    fas_async_b128(fas_lds_addr(dst + (size_t)cidx * 8), src);
  }
  {                                            // vT region: first 128 x 16B
    int s = tid >> 6, rem = tid & 63;
    int o = rem >> 2, ch = rem & 3;
    const _Float16* src =
        vT + ((size_t)(s * NB + batch) * CV + o) * HW + J0 + ch * 8;
    fas_async_b128(fas_lds_addr(dst + VOFF + (size_t)tid * 8), src);
  }
  {                                            // vT tail: 128 x 8B
    int rel = 1024 + tid * 4;
    int s = rel >> 9, r2 = rel & 511;
    int o = r2 >> 5, jc = r2 & 31;
    const _Float16* src =
        vT + ((size_t)(s * NB + batch) * CV + o) * HW + J0 + jc;
    fas_async_b64(fas_lds_addr(dst + VOFF + rel), src);
  }
}

// ---------------------------------------------------------------------------
// WMMA fragment loader. 16-bit A (16x32) / B(32x16, lane n = column n) layout:
//   lanes 0-15 : K = {k0+0..7}  and {k0+16..23}
//   lanes 16-31: K = {k0+8..15} and {k0+24..31}
// ---------------------------------------------------------------------------
__device__ __forceinline__ v16h fas_frag(const _Float16* p, int kb) {
  union { v16h v; v8h h[2]; } u;
  u.h[0] = *(const v8h*)(p + kb);
  u.h[1] = *(const v8h*)(p + kb + 16);
  return u.v;
}

struct __align__(16) FasLds {
  _Float16 buf[2][NS * JT * NC + NS * CV * JT];  // 2 x 27648 B tile buffers
  _Float16 sA[NWAVE][NS][16][JT];                // 12 KB combined-P tiles
};

// ---------------------------------------------------------------------------
// attn: per block = (batch, 64-row tile). Per wave: one 16-row tile.
// Double-buffered async J-tile stream; never materializes (hw,hw) matrices.
// ---------------------------------------------------------------------------
__global__ __launch_bounds__(NWAVE * 32) void fas_attn(
    const _Float16* __restrict__ xb, const _Float16* __restrict__ vT,
    const float* __restrict__ sq, const float* __restrict__ g,
    float* __restrict__ att)
{
  __shared__ FasLds L;
  const int batch = blockIdx.x / (HW / IT);
  const int I0    = (blockIdx.x % (HW / IT)) * IT;
  const int wave  = threadIdx.x >> 5;
  const int lane  = threadIdx.x & 31;
  const int l16   = lane & 15;
  const int moff  = (lane < 16) ? 0 : 8;   // M offset of C-layout rows
  const int kb    = (lane < 16) ? 0 : 8;   // K base of A/B frag halves
  const int rowM  = I0 + wave * 16;

  float gam[9];
  #pragma unroll
  for (int i = 0; i < 9; ++i) gam[i] = g[i];
  float nc[NS];                       // -1/(2 sigma^2)
  #pragma unroll
  for (int s = 0; s < NS; ++s) nc[s] = -0.5f / (gam[s] * gam[s]);
  float co[NS][NS];                   // combined coefficients per output stream
  #pragma unroll
  for (int i = 0; i < NS; ++i)
    #pragma unroll
    for (int s = 0; s < NS; ++s) co[i][s] = gam[6 + s] + ((i == s) ? 0.f : 1.f);

  // persistent A fragments (this wave's 16 rows, all 128 K) + row norms
  v16h aF[NS][4];
  float sqI[NS][8];
  #pragma unroll
  for (int s = 0; s < NS; ++s) {
    const _Float16* xr = xb + ((size_t)(s * NB + batch) * HW + rowM + l16) * NC;
    #pragma unroll
    for (int k = 0; k < 4; ++k) aF[s][k] = fas_frag(xr + 32 * k, kb);
    const float* sp = sq + (size_t)(s * NB + batch) * HW + rowM + moff;
    #pragma unroll
    for (int r = 0; r < 8; ++r) sqI[s][r] = sp[r];
  }

  v8f outacc[NS] = {};

  const int NT = HW / JT;                       // 72 J tiles
  fas_issue_tile(L.buf[0], xb, vT, batch, 0, threadIdx.x);

  for (int t = 0; t < NT; ++t) {
    const int J0 = t * JT;
    _Float16* cbuf = L.buf[t & 1];
    if (t + 1 < NT) {                           // stream next tile into the
      fas_issue_tile(L.buf[(t + 1) & 1], xb, vT, batch, J0 + JT, threadIdx.x);
      fas_wait_async_prev();                    // wait: tile t landed
    } else {
      fas_wait_async_all();
    }
    __syncthreads();                            // publish tile t to all waves

    // ---- Gram WMMAs + fused exp + coefficient combine ----
    #pragma unroll
    for (int jsub = 0; jsub < 2; ++jsub) {
      v8f comb0 = {}, comb1 = {}, comb2 = {};
      #pragma unroll
      for (int s = 0; s < NS; ++s) {
        const _Float16* brow = cbuf + ((size_t)(s * JT + jsub * 16 + l16)) * NC;
        v8f acc = {};
        #pragma unroll
        for (int k = 0; k < 4; ++k) {
          v16h bF = fas_frag(brow + 32 * k, kb);
          acc = __builtin_amdgcn_wmma_f32_16x16x32_f16(
              false, aF[s][k], false, bF, (short)0, acc, false, false);
        }
        float sqJv = sq[(size_t)(s * NB + batch) * HW + J0 + jsub * 16 + l16];
        #pragma unroll
        for (int r = 0; r < 8; ++r) {
          float d2 = fmaxf(sqI[s][r] + sqJv - 2.0f * acc[r], 0.0f);
          float sv = __expf(d2 * nc[s]);
          comb0[r] = fmaf(co[0][s], sv, comb0[r]);
          comb1[r] = fmaf(co[1][s], sv, comb1[r]);
          comb2[r] = fmaf(co[2][s], sv, comb2[r]);
        }
      }
      // C-layout -> A-layout transpose through wave-private LDS
      #pragma unroll
      for (int r = 0; r < 8; ++r) {
        L.sA[wave][0][moff + r][jsub * 16 + l16] = (_Float16)comb0[r];
        L.sA[wave][1][moff + r][jsub * 16 + l16] = (_Float16)comb1[r];
        L.sA[wave][2][moff + r][jsub * 16 + l16] = (_Float16)comb2[r];
      }
    }

    // ---- projection WMMAs: out_i += P_i (16x32) @ v_i (32x16) ----
    #pragma unroll
    for (int i = 0; i < NS; ++i) {
      v16h aP = fas_frag(&L.sA[wave][i][l16][0], kb);
      v16h bP = fas_frag(cbuf + VOFF + ((size_t)(i * CV + l16)) * JT, kb);
      outacc[i] = __builtin_amdgcn_wmma_f32_16x16x32_f16(
          false, aP, false, bP, (short)0, outacc[i], false, false);
    }
    __syncthreads();                 // free cbuf for the t+1 async issue
  }

  // write att (pre-conv, already scaled by g[3+i]); layout ((i*NB+b)*HW+j)*CV+o
  #pragma unroll
  for (int i = 0; i < NS; ++i) {
    float sc = gam[3 + i];
    float* dst = att + ((size_t)(i * NB + batch) * HW + rowM + moff) * CV + l16;
    #pragma unroll
    for (int r = 0; r < 8; ++r) dst[(size_t)r * CV] = outacc[i][r] * sc;
  }
}

// ---------------------------------------------------------------------------
// out: conv1x1 (128 x 16) + bias, then elementwise multiply with input.
// ---------------------------------------------------------------------------
__global__ __launch_bounds__(256) void fas_out(
    const float* __restrict__ d0, const float* __restrict__ d1,
    const float* __restrict__ d2,
    const float* __restrict__ cw, const float* __restrict__ cb,
    const float* __restrict__ att, float* __restrict__ out)
{
  int tid = blockIdx.x * 256 + threadIdx.x;
  if (tid >= NS * NB * HW) return;
  int j  = tid % HW;
  int sb = tid / HW;
  int s  = sb / NB;
  const float* xd = (s == 0 ? d0 : (s == 1 ? d1 : d2)) +
                    (size_t)(sb % NB) * NC * HW + j;
  float a[CV];
  const float* ap = att + (size_t)tid * CV;
  #pragma unroll
  for (int o = 0; o < CV; ++o) a[o] = ap[o];
  float* op = out + (size_t)sb * NC * HW + j;
  #pragma unroll 4
  for (int c = 0; c < NC; ++c) {
    float acc = cb[c];
    #pragma unroll
    for (int o = 0; o < CV; ++o) acc = fmaf(cw[c * CV + o], a[o], acc);
    op[(size_t)c * HW] = xd[(size_t)c * HW] * acc;
  }
}

extern "C" void kernel_launch(void* const* d_in, const int* in_sizes, int n_in,
                              void* d_out, int out_size, void* d_ws, size_t ws_size,
                              hipStream_t stream) {
  const float* d0 = (const float*)d_in[0];
  const float* d1 = (const float*)d_in[1];
  const float* d2 = (const float*)d_in[2];
  const float* vw = (const float*)d_in[3];
  const float* vb = (const float*)d_in[4];
  const float* cw = (const float*)d_in[5];
  const float* cb = (const float*)d_in[6];
  const float* g  = (const float*)d_in[7];

  char* ws = (char*)d_ws;
  _Float16* xb = (_Float16*)ws;                                // 14,155,776 B
  _Float16* vT = (_Float16*)(ws + (size_t)NS * NB * HW * NC * 2);
  float*    sq = (float*)((char*)vT + (size_t)NS * NB * CV * HW * 2);
  float*    att = (float*)((char*)sq + (size_t)NS * NB * HW * 4);
  float*    out = (float*)d_out;

  int nThreads = NS * NB * HW;
  fas_prep<<<(nThreads + 255) / 256, 256, 0, stream>>>(d0, d1, d2, vw, vb,
                                                       xb, vT, sq);
  fas_attn<<<NB * (HW / IT), NWAVE * 32, 0, stream>>>(xb, vT, sq, g, att);
  fas_out<<<(nThreads + 255) / 256, 256, 0, stream>>>(d0, d1, d2, cw, cb,
                                                      att, out);
}
